// EfficientAttention_32959579030330
// MI455X (gfx1250) — matrix operand order repaired and checked by reading, not compiled, not run on hardware
//
#include <hip/hip_runtime.h>
#include <hip/hip_bf16.h>
#include <stdint.h>

// ---------------------------------------------------------------------------
// EfficientAttention (linear attention) for MI455X / gfx1250.
// bf16 WMMA everywhere matrix math happens; TDM (tensor_load_to_lds) feeds
// the big GEMMs with hardware-padded LDS tiles, double-buffered.
// Shapes: Lq=Lk=4096, N=4, E=1024, H=16, D=64, M = Lq*N = 16384 rows.
// ---------------------------------------------------------------------------

typedef __attribute__((ext_vector_type(16))) __bf16 v16bf;
typedef __attribute__((ext_vector_type(8)))  __bf16 v8bf;
typedef __attribute__((ext_vector_type(4)))  __bf16 v4bf;
typedef __attribute__((ext_vector_type(8)))  float  v8f;
typedef __attribute__((ext_vector_type(4)))  float  v4f;
typedef __attribute__((ext_vector_type(4)))  unsigned int u32x4;
typedef __attribute__((ext_vector_type(8)))  int    i32x8;
typedef __attribute__((ext_vector_type(4)))  int    i32x4;

#define EMBED 1024
#define NHEAD 16
#define HDIM  64
#define NBAT  4
#define LSEQ  4096
#define MROWS (LSEQ * NBAT)   // 16384
#define NECOL (NBAT * EMBED)  // 4096 columns for the L-softmax

#define HAVE_TDM __has_builtin(__builtin_amdgcn_tensor_load_to_lds)

static __device__ __forceinline__ __bf16 f2bf(float f) {
  union { float f; uint32_t u; } a; a.f = f;
  uint32_t r = (a.u + 0x7FFFu + ((a.u >> 16) & 1u)) >> 16;   // RNE
  union { uint16_t s; __bf16 b; } o; o.s = (uint16_t)r;
  return o.b;
}

static __device__ __forceinline__ v16bf mk16(v8bf lo, v8bf hi) {
  v16bf r;
#pragma unroll
  for (int i = 0; i < 8; ++i) { r[i] = lo[i]; r[i + 8] = hi[i]; }
  return r;
}

static __device__ __forceinline__ v8f wmma_bf16(v16bf a, v16bf b, v8f c) {
  return __builtin_amdgcn_wmma_f32_16x16x32_bf16(false, a, false, b,
                                                 (short)0, c, false, false);
}

// Generic LDS pointer -> 32-bit LDS offset (ISA: LDS aperture addr[31:0]).
static __device__ __forceinline__ unsigned lds_addr_of(const void* p) {
  return (unsigned)(unsigned long long)p;
}

#if HAVE_TDM
// ---------------------------------------------------------------------------
// Issue one TDM 2-D tile load: global (bf16, row stride = row_stride_elems)
// -> LDS at lds_byte_addr, tile tile_w x tile_h, with hardware row padding:
// pad_interval=3 (16 dwords = 64B = 32 bf16), pad_amount=3 (4 dwords = 16B)
// => LDS row pitch 80B = 40 bf16, matching the [..][40] shared tiles below.
// D# layout per CDNA5 ISA 8.3/8.4 (group0 128b, group1 256b).
// ---------------------------------------------------------------------------
static __device__ __forceinline__ void tdm_load_2d(
    unsigned lds_byte_addr, const void* gptr,
    unsigned tile_w, unsigned tile_h,
    unsigned row_stride_elems, unsigned tensor_w, unsigned tensor_h) {
  unsigned long long ga = (unsigned long long)gptr;
  u32x4 g0;
  g0[0] = 1u;                                               // count=1 (user)
  g0[1] = lds_byte_addr;                                    // lds_addr
  g0[2] = (unsigned)ga;                                     // global_addr lo
  g0[3] = (unsigned)((ga >> 32) & 0x1FFFFFFu) | (2u << 30); // addr hi | type=2
  i32x8 g1;
  g1[0] = (int)((1u << 16)      // data_size = 2 bytes (bf16)
              | (1u << 20)      // pad_enable
              | (3u << 22)      // pad_interval: 16 dwords
              | (3u << 25));    // pad_amount: 4 dwords
  g1[1] = (int)((tensor_w & 0xFFFFu) << 16);                        // dim0 lo16
  g1[2] = (int)((tensor_w >> 16) | ((tensor_h & 0xFFFFu) << 16));   // dim0hi|dim1lo
  g1[3] = (int)((tensor_h >> 16) | (tile_w << 16));                 // dim1hi|tile0
  g1[4] = (int)tile_h;                                              // tile1 (tile2=0)
  g1[5] = (int)row_stride_elems;                                    // stride0 lo32
  g1[6] = 0;                                                        // stride0 hi
  g1[7] = 0;
  i32x4 z4 = {};
#if __clang_major__ >= 23
  i32x8 z8 = {};
  __builtin_amdgcn_tensor_load_to_lds(g0, g1, z4, z4, z8, 0);
#else
  __builtin_amdgcn_tensor_load_to_lds(g0, g1, z4, z4, 0);
#endif
}
#endif  // HAVE_TDM

// ---------------------------------------------------------------------------
// fp32 -> bf16 conversion (one float4 -> one bf16x4 per thread).
// ---------------------------------------------------------------------------
__global__ __launch_bounds__(256) void conv_f32_bf16(
    const float* __restrict__ src, __bf16* __restrict__ dst, int n4) {
  int i = blockIdx.x * 256 + threadIdx.x;
  if (i < n4) {
    v4f x = ((const v4f*)src)[i];
    v4bf o;
#pragma unroll
    for (int j = 0; j < 4; ++j) o[j] = f2bf(x[j]);
    ((v4bf*)dst)[i] = o;
  }
}

// ---------------------------------------------------------------------------
// C[M,Nc] = A[M,K](bf16) @ W[Nc,K]^T (bf16) + bias.  (torch Linear layout)
// Block tile 128x64, 8 waves of 32x32, BK=32.
// TDM path: wave 0 issues two tensor_load_to_lds per K-step into the inactive
// LDS buffer (double buffered), s_wait_tensorcnt + barrier publishes it; the
// other 7 waves spend zero instructions on data movement.
// Output: fp32 (Cf != null) or bf16 (Cb).
// ---------------------------------------------------------------------------
__global__ __launch_bounds__(256) void gemm_bf16(
    const __bf16* __restrict__ A, const __bf16* __restrict__ W,
    const float* __restrict__ bias,
    float* __restrict__ Cf, __bf16* __restrict__ Cb,
    int M, int Nc, int K) {
  __shared__ __bf16 As[2][128][40];   // 80B row pitch (TDM pad / manual pad)
  __shared__ __bf16 Bs[2][64][40];

  const int tid  = threadIdx.x;
  const int lane = tid & 31;
  const int wave = tid >> 5;
  const int wm   = wave & 3;   // 4 wave rows
  const int wn   = wave >> 2;  // 2 wave cols
  const int bm   = blockIdx.x * 128;
  const int bn   = blockIdx.y * 64;

  const int arow  = lane & 15;
  const int kb    = (lane >> 4) * 8;    // A-fragment K sub-base
  const int khalf = (lane >> 4) * 16;   // unused; B reads at kb, kb + 16 like A

  v8f acc[2][2] = {};

#if HAVE_TDM
  const bool issuer = (wave == 0);
  if (issuer) {
    tdm_load_2d(lds_addr_of(&As[0][0][0]), A + (size_t)bm * K, 32, 128, K, K, M);
    tdm_load_2d(lds_addr_of(&Bs[0][0][0]), W + (size_t)bn * K, 32, 64,  K, K, Nc);
    __builtin_amdgcn_s_wait_tensorcnt(0);
  }
  __syncthreads();

  int cur = 0;
  for (int kk = 0; kk < K; kk += 32) {
    const int nxt = cur ^ 1;
    const bool more = (kk + 32 < K);
    if (issuer && more) {
      tdm_load_2d(lds_addr_of(&As[nxt][0][0]),
                  A + (size_t)bm * K + kk + 32, 32, 128, K, K, M);
      tdm_load_2d(lds_addr_of(&Bs[nxt][0][0]),
                  W + (size_t)bn * K + kk + 32, 32, 64,  K, K, Nc);
    }

    v16bf bfr[2];
#pragma unroll
    for (int jn = 0; jn < 2; ++jn) {
      int cn = wn * 32 + jn * 16 + arow;
      bfr[jn] = mk16(*(const v8bf*)&Bs[cur][cn][kb],
                     *(const v8bf*)&Bs[cur][cn][kb + 16]);
    }
#pragma unroll
    for (int im = 0; im < 2; ++im) {
      int r = wm * 32 + im * 16 + arow;
      v16bf a = mk16(*(const v8bf*)&As[cur][r][kb],
                     *(const v8bf*)&As[cur][r][kb + 16]);
#pragma unroll
      for (int jn = 0; jn < 2; ++jn)
        acc[im][jn] = wmma_bf16(a, bfr[jn], acc[im][jn]);
    }

    if (more) {
      if (issuer) __builtin_amdgcn_s_wait_tensorcnt(0);
      __syncthreads();
      cur = nxt;
    }
  }
#else
  // Fallback: VGPR-staged loads (no TDM builtin on this toolchain / host pass)
  for (int kk = 0; kk < K; kk += 32) {
#pragma unroll
    for (int t = 0; t < 2; ++t) {
      int c  = tid * 2 + t;
      int r  = c >> 2;
      int c8 = (c & 3) * 8;
      *(v8bf*)&As[0][r][c8] = *(const v8bf*)&A[(size_t)(bm + r) * K + kk + c8];
    }
    {
      int r  = tid >> 2;
      int c8 = (tid & 3) * 8;
      *(v8bf*)&Bs[0][r][c8] = *(const v8bf*)&W[(size_t)(bn + r) * K + kk + c8];
    }
    if (kk + 32 < K) {
      __builtin_prefetch(&A[(size_t)(bm + (tid >> 1)) * K + kk + 32], 0, 1);
      __builtin_prefetch(&W[(size_t)(bn + (tid >> 2)) * K + kk + 32], 0, 1);
    }
    __syncthreads();
    v16bf bfr[2];
#pragma unroll
    for (int jn = 0; jn < 2; ++jn) {
      int cn = wn * 32 + jn * 16 + arow;
      bfr[jn] = mk16(*(const v8bf*)&Bs[0][cn][kb],
                     *(const v8bf*)&Bs[0][cn][kb + 16]);
    }
#pragma unroll
    for (int im = 0; im < 2; ++im) {
      int r = wm * 32 + im * 16 + arow;
      v16bf a = mk16(*(const v8bf*)&As[0][r][kb],
                     *(const v8bf*)&As[0][r][kb + 16]);
#pragma unroll
      for (int jn = 0; jn < 2; ++jn)
        acc[im][jn] = wmma_bf16(a, bfr[jn], acc[im][jn]);
    }
    __syncthreads();
  }
#endif

  // Epilogue: C layout — VGPR i holds M = i + 8*(lane>=16), N = lane&15.
#pragma unroll
  for (int jn = 0; jn < 2; ++jn) {
    int col = bn + wn * 32 + jn * 16 + (lane & 15);
    float bv = bias ? bias[col] : 0.0f;
#pragma unroll
    for (int im = 0; im < 2; ++im) {
#pragma unroll
      for (int i = 0; i < 8; ++i) {
        int row = bm + wm * 32 + im * 16 + i + ((lane >> 4) * 8);
        float vvv = acc[im][jn][i] + bv;
        if (Cf) Cf[(size_t)row * Nc + col] = vvv;
        else    Cb[(size_t)row * Nc + col] = f2bf(vvv);
      }
    }
  }
}

// ---------------------------------------------------------------------------
// K softmax over the sequence axis (per column of kp viewed as [L, N*E]).
// Split-L partials -> merge -> normalize, all coalesced across columns.
// ---------------------------------------------------------------------------
__global__ __launch_bounds__(256) void ksm_partial(
    const float* __restrict__ kp, float* __restrict__ pm, float* __restrict__ ps) {
  int col = blockIdx.x * 256 + threadIdx.x;   // 0..4095
  int lch = blockIdx.y;                       // 0..15, 256 rows each
  const float* p = kp + (size_t)lch * 256 * NECOL + col;
  float m = -3.4e38f;
  for (int l = 0; l < 256; ++l) m = fmaxf(m, p[(size_t)l * NECOL]);
  float s = 0.0f;
  for (int l = 0; l < 256; ++l) s += __expf(p[(size_t)l * NECOL] - m);
  pm[lch * NECOL + col] = m;
  ps[lch * NECOL + col] = s;
}

__global__ __launch_bounds__(256) void ksm_reduce(
    const float* __restrict__ pm, const float* __restrict__ ps,
    float* __restrict__ cm, float* __restrict__ cs) {
  int col = blockIdx.x * 256 + threadIdx.x;
  float m = -3.4e38f;
#pragma unroll
  for (int i = 0; i < 16; ++i) m = fmaxf(m, pm[i * NECOL + col]);
  float s = 0.0f;
#pragma unroll
  for (int i = 0; i < 16; ++i) s += ps[i * NECOL + col] * __expf(pm[i * NECOL + col] - m);
  cm[col] = m;
  cs[col] = s;
}

__global__ __launch_bounds__(256) void ksm_norm(
    const float* __restrict__ kp, const float* __restrict__ cm,
    const float* __restrict__ cs, __bf16* __restrict__ kh) {
  int col = blockIdx.x * 256 + threadIdx.x;
  int lch = blockIdx.y;
  float m  = cm[col];
  float rs = 1.0f / cs[col];
  const float* p = kp + (size_t)lch * 256 * NECOL + col;
  __bf16*      o = kh + (size_t)lch * 256 * NECOL + col;
  for (int l = 0; l < 256; ++l)
    o[(size_t)l * NECOL] = f2bf(__expf(p[(size_t)l * NECOL] - m) * rs);
}

// ---------------------------------------------------------------------------
// Q softmax over the feature axis (per head, 64 values). One wave per
// 1024-wide row; each lane owns 32 contiguous floats = half a head; pairwise
// reduce with shfl_xor(1).
// ---------------------------------------------------------------------------
__global__ __launch_bounds__(256) void qsm_kernel(
    const float* __restrict__ qp, __bf16* __restrict__ qh) {
  int row  = blockIdx.x * 8 + (threadIdx.x >> 5);
  int lane = threadIdx.x & 31;
  const float* p = qp + (size_t)row * EMBED + lane * 32;
  float x[32];
#pragma unroll
  for (int j = 0; j < 8; ++j) {
    v4f t = ((const v4f*)p)[j];
#pragma unroll
    for (int q4 = 0; q4 < 4; ++q4) x[j * 4 + q4] = t[q4];
  }
  float m = -3.4e38f;
#pragma unroll
  for (int j = 0; j < 32; ++j) m = fmaxf(m, x[j]);
  m = fmaxf(m, __shfl_xor(m, 1, 32));
  float s = 0.0f;
#pragma unroll
  for (int j = 0; j < 32; ++j) { x[j] = __expf(x[j] - m); s += x[j]; }
  s += __shfl_xor(s, 1, 32);
  float rs = 1.0f / s;
  __bf16* o = qh + (size_t)row * EMBED + lane * 32;
#pragma unroll
  for (int j = 0; j < 4; ++j) {
    v8bf ov;
#pragma unroll
    for (int q8 = 0; q8 < 8; ++q8) ov[q8] = f2bf(x[j * 8 + q8] * rs);
    ((v8bf*)o)[j] = ov;
  }
}

// ---------------------------------------------------------------------------
// v_prime[n,h] = kh[n,h]^T @ vp[n,h]  — [64x4096] @ [4096x64] = [64x64].
// One block per (n,h), 4 waves each owning a 16x64 output strip. K/V tiles
// are written transposed into LDS so both WMMA operands read contiguous K.
// ---------------------------------------------------------------------------
__global__ __launch_bounds__(128) void vprime_kernel(
    const __bf16* __restrict__ kh, const __bf16* __restrict__ vp,
    __bf16* __restrict__ vpr) {
  __shared__ __bf16 kT[64][40];
  __shared__ __bf16 vT[64][40];
  const int n = blockIdx.x >> 4, h = blockIdx.x & 15;
  const int tid = threadIdx.x, lane = tid & 31, wave = tid >> 5;
  const size_t hoff = (size_t)n * EMBED + h * HDIM;
  const int arow = lane & 15, kb = (lane >> 4) * 8, khalf = (lane >> 4) * 16;
  v8f acc[4] = {};

  for (int kc = 0; kc < 128; ++kc) {
    int lbase = kc * 32;
#pragma unroll
    for (int t = 0; t < 2; ++t) {
      int c  = tid * 2 + t;        // 0..255
      int l  = c >> 3;             // 0..31
      int e8 = (c & 7) * 8;
      v8bf kv = *(const v8bf*)&kh[(size_t)(lbase + l) * NECOL + hoff + e8];
      v8bf vv = *(const v8bf*)&vp[(size_t)(lbase + l) * NECOL + hoff + e8];
#pragma unroll
      for (int j = 0; j < 8; ++j) { kT[e8 + j][l] = kv[j]; vT[e8 + j][l] = vv[j]; }
    }
    __syncthreads();

    int dbase = wave * 16;
    v16bf a = mk16(*(const v8bf*)&kT[dbase + arow][kb],
                   *(const v8bf*)&kT[dbase + arow][kb + 16]);
#pragma unroll
    for (int j = 0; j < 4; ++j) {
      v16bf b = mk16(*(const v8bf*)&vT[j * 16 + arow][kb],
                     *(const v8bf*)&vT[j * 16 + arow][kb + 16]);
      acc[j] = wmma_bf16(a, b, acc[j]);
    }
    __syncthreads();
  }

  const size_t base = (size_t)(n * NHEAD + h) * HDIM * HDIM;
#pragma unroll
  for (int j = 0; j < 4; ++j)
#pragma unroll
    for (int i = 0; i < 8; ++i) {
      int d = wave * 16 + i + ((lane >> 4) * 8);
      int e = j * 16 + (lane & 15);
      vpr[base + (size_t)d * HDIM + e] = f2bf(acc[j][i]);
    }
}

// ---------------------------------------------------------------------------
// attn[n,h] = qh[n,h] @ v_prime[n,h]  — [4096x64] @ [64x64].
// Block = (128 rows of one (n,h)); v_prime lives transposed in LDS for the
// whole block; A fragments stream straight from global (K contiguous).
// ---------------------------------------------------------------------------
__global__ __launch_bounds__(256) void attn_kernel(
    const __bf16* __restrict__ qh, const __bf16* __restrict__ vpr,
    __bf16* __restrict__ att) {
  __shared__ __bf16 vpT[64][72];   // [e][d], 144B rows (16B-aligned)
  const int n = blockIdx.y >> 4, h = blockIdx.y & 15;
  const int tid = threadIdx.x, lane = tid & 31, wave = tid >> 5;
  const size_t base = (size_t)(n * NHEAD + h) * HDIM * HDIM;

#pragma unroll
  for (int t = 0; t < 2; ++t) {
    int c  = tid * 2 + t;          // 0..511
    int d  = c >> 3;               // 0..63
    int e8 = (c & 7) * 8;
    v8bf v = *(const v8bf*)&vpr[base + (size_t)d * HDIM + e8];
#pragma unroll
    for (int j = 0; j < 8; ++j) vpT[e8 + j][d] = v[j];
  }
  __syncthreads();

  const int lbase = blockIdx.x * 128 + wave * 16;
  const int arow = lane & 15, kb = (lane >> 4) * 8, khalf = (lane >> 4) * 16;
  v8f acc[4] = {};

#pragma unroll
  for (int ks = 0; ks < 2; ++ks) {
    int k0 = ks * 32;
    size_t qa = ((size_t)(lbase + arow) * NBAT + n) * EMBED + h * HDIM + k0 + kb;
    v16bf a = mk16(*(const v8bf*)&qh[qa], *(const v8bf*)&qh[qa + 16]);
#pragma unroll
    for (int j = 0; j < 4; ++j) {
      v16bf b = mk16(*(const v8bf*)&vpT[j * 16 + arow][k0 + kb],
                     *(const v8bf*)&vpT[j * 16 + arow][k0 + kb + 16]);
      acc[j] = wmma_bf16(a, b, acc[j]);
    }
  }

#pragma unroll
  for (int j = 0; j < 4; ++j)
#pragma unroll
    for (int i = 0; i < 8; ++i) {
      int l = lbase + i + ((lane >> 4) * 8);
      int e = j * 16 + (lane & 15);
      att[((size_t)l * NBAT + n) * EMBED + h * HDIM + e] = f2bf(acc[j][i]);
    }
}

// ---------------------------------------------------------------------------
// Host launcher.
// ---------------------------------------------------------------------------
extern "C" void kernel_launch(void* const* d_in, const int* in_sizes, int n_in,
                              void* d_out, int out_size, void* d_ws, size_t ws_size,
                              hipStream_t stream) {
  const float* q  = (const float*)d_in[0];
  const float* k  = (const float*)d_in[1];
  const float* v  = (const float*)d_in[2];
  const float* Wq = (const float*)d_in[3];
  const float* bq = (const float*)d_in[4];
  const float* Wk = (const float*)d_in[5];
  const float* bk = (const float*)d_in[6];
  const float* Wv = (const float*)d_in[7];
  const float* bv = (const float*)d_in[8];
  const float* Wo = (const float*)d_in[9];
  const float* bo = (const float*)d_in[10];
  float* out = (float*)d_out;

  char* ws = (char*)d_ws;
  const size_t MB = 1ull << 20;
  __bf16* Xq  = (__bf16*)(ws + 0 * MB);     // 32 MB each
  __bf16* Xk  = (__bf16*)(ws + 32 * MB);
  __bf16* Xv  = (__bf16*)(ws + 64 * MB);
  __bf16* WqB = (__bf16*)(ws + 96 * MB);    // 2 MB each
  __bf16* WkB = (__bf16*)(ws + 98 * MB);
  __bf16* WvB = (__bf16*)(ws + 100 * MB);
  __bf16* WoB = (__bf16*)(ws + 102 * MB);
  float*  qp  = (float*)(ws + 104 * MB);    // 64 MB
  float*  kp  = (float*)(ws + 168 * MB);    // 64 MB
  __bf16* vpb = (__bf16*)(ws + 232 * MB);   // 32 MB
  __bf16* khb = (__bf16*)(ws + 264 * MB);   // 32 MB
  __bf16* qhb = (__bf16*)(ws + 296 * MB);   // 32 MB
  float*  pm  = (float*)(ws + 328 * MB);            // 16*4096 f32
  float*  ps  = (float*)(ws + 328 * MB + 256 * 1024);
  float*  cm  = (float*)(ws + 328 * MB + 512 * 1024);
  float*  cs  = (float*)(ws + 328 * MB + 528 * 1024);
  __bf16* vpr = (__bf16*)(ws + 329 * MB);   // 512 KB
  __bf16* atb = (__bf16*)(ws + 330 * MB);   // 32 MB

  const int nQKV4 = MROWS * EMBED / 4;  // 4,194,304
  const int nW4   = EMBED * EMBED / 4;  // 262,144
  conv_f32_bf16<<<nQKV4 / 256, 256, 0, stream>>>(q,  Xq,  nQKV4);
  conv_f32_bf16<<<nQKV4 / 256, 256, 0, stream>>>(k,  Xk,  nQKV4);
  conv_f32_bf16<<<nQKV4 / 256, 256, 0, stream>>>(v,  Xv,  nQKV4);
  conv_f32_bf16<<<nW4 / 256,  256, 0, stream>>>(Wq, WqB, nW4);
  conv_f32_bf16<<<nW4 / 256,  256, 0, stream>>>(Wk, WkB, nW4);
  conv_f32_bf16<<<nW4 / 256,  256, 0, stream>>>(Wv, WvB, nW4);
  conv_f32_bf16<<<nW4 / 256,  256, 0, stream>>>(Wo, WoB, nW4);

  dim3 ggrid(MROWS / 128, EMBED / 64);  // 128 x 16
  gemm_bf16<<<ggrid, 256, 0, stream>>>(Xq, WqB, bq, qp, nullptr, MROWS, EMBED, EMBED);
  gemm_bf16<<<ggrid, 256, 0, stream>>>(Xk, WkB, bk, kp, nullptr, MROWS, EMBED, EMBED);
  gemm_bf16<<<ggrid, 256, 0, stream>>>(Xv, WvB, bv, nullptr, vpb, MROWS, EMBED, EMBED);

  ksm_partial<<<dim3(NECOL / 256, 16), 256, 0, stream>>>(kp, pm, ps);
  ksm_reduce<<<NECOL / 256, 256, 0, stream>>>(pm, ps, cm, cs);
  ksm_norm<<<dim3(NECOL / 256, 16), 256, 0, stream>>>(kp, cm, cs, khb);

  qsm_kernel<<<MROWS / 8, 256, 0, stream>>>(qp, qhb);

  vprime_kernel<<<NBAT * NHEAD, 128, 0, stream>>>(khb, vpb, vpr);
  attn_kernel<<<dim3(LSEQ / 128, NBAT * NHEAD), 256, 0, stream>>>(qhb, vpr, atb);

  gemm_bf16<<<ggrid, 256, 0, stream>>>(atb, WoB, bo, out, nullptr, MROWS, EMBED, EMBED);
}
